// DGDN_41480794144816
// MI455X (gfx1250) — matrix operand order repaired and checked by reading, not compiled, not run on hardware
//
#include <hip/hip_runtime.h>
#include <hip/hip_bf16.h>
#include <stdint.h>

// ---------------- CDNA5 WMMA types ----------------
typedef __attribute__((ext_vector_type(8)))  _Float16 v8h;
typedef __attribute__((ext_vector_type(16))) _Float16 v16h;
typedef __attribute__((ext_vector_type(8)))  float    v8f;
typedef __attribute__((ext_vector_type(4)))  float    v4f;

#define WMMA_F32_F16(A, B, C) \
  __builtin_amdgcn_wmma_f32_16x16x32_f16(false, (A), false, (B), (short)0, (C), false, false)

// ISA 7.12.2 fragment loaders: each is exactly 2x ds_load_b128 (unconditional).
// A-matrix 16x32 f16: elem e<8 -> K = kb + 8*half + e ; e>=8 -> K = kb + 16 + 8*half + (e-8)
__device__ __forceinline__ v16h ldA(const _Float16* base, int ldk, int m, int kb, int half) {
  const _Float16* p = base + (size_t)m * ldk + kb;
  v8h lo = *(const v8h*)(p + 8 * half);
  v8h hi = *(const v8h*)(p + 16 + 8 * half);
  return __builtin_shufflevector(lo, hi, 0,1,2,3,4,5,6,7,8,9,10,11,12,13,14,15);
}
// B-matrix 32x16 f16: elem e<8 -> K = kb + 8*half + e ; e>=8 -> K = kb + 16 + 8*half + (e-8)
__device__ __forceinline__ v16h ldB(const _Float16* base, int ldk, int n, int kb, int half) {
  const _Float16* p = base + (size_t)n * ldk + kb;
  v8h lo = *(const v8h*)(p + 8 * half);
  v8h hi = *(const v8h*)(p + 16 + 8 * half);
  return __builtin_shufflevector(lo, hi, 0,1,2,3,4,5,6,7,8,9,10,11,12,13,14,15);
}

// ---------------- deterministic counter RNG ----------------
__device__ __forceinline__ uint32_t hash3(uint32_t a, uint32_t b, uint32_t c) {
  uint32_t h = a * 0x9E3779B1u;
  h ^= (b * 0x85EBCA77u) + 0x9E3779B9u + (h << 6) + (h >> 2);
  h ^= (c * 0xC2B2AE3Du) + 0x9E3779B9u + (h << 6) + (h >> 2);
  h ^= h >> 16; h *= 0x7FEB352Du; h ^= h >> 15; h *= 0x846CA68Bu; h ^= h >> 16;
  return h;
}
__device__ __forceinline__ float u01(uint32_t h) {
  return (float)(h >> 8) * (1.0f / 16777216.0f) + (0.5f / 16777216.0f);
}
__device__ __forceinline__ float gaussf(uint32_t a, uint32_t b, uint32_t c) {
  float u1 = u01(hash3(a, b, c));
  float u2 = u01(hash3(a, b, c ^ 0x5BD1E99u));
  return __builtin_sqrtf(-2.0f * __logf(u1)) * __cosf(6.28318530718f * u2);
}

#define BATCH 4096
#define K1C 30
#define K2C 80

extern __shared__ __align__(16) unsigned char smem_raw[];

// =====================================================================
// Kernel 1: enc1 conv8x8 (implicit GEMM M=441->448, N=30->32, K=64) +
// stochastic pooling. Per-image block; im2col A panel + c1 live in LDS.
// Dynamic LDS: As 57344 | c1 57344 | Bs 4096 | xs 3136 | pw 648 = 122568 B
// =====================================================================
__global__ __launch_bounds__(256) void enc1_pool_kernel(
    const float* __restrict__ x, const float* __restrict__ enc1_w,
    const float* __restrict__ pw1, const float* __restrict__ pw2,
    float* __restrict__ pooled, float* __restrict__ eta_out) {
  _Float16* As = (_Float16*)smem_raw;                            // [448][64]
  float*    c1 = (float*)(smem_raw + 57344);                     // [448][32]
  _Float16* Bs = (_Float16*)(smem_raw + 57344 + 57344);          // [32][64]
  float*    xs = (float*)(smem_raw + 57344 + 57344 + 4096);      // [784]
  float*    pw = xs + 784;                                       // [162]

  const int b = blockIdx.x, tid = threadIdx.x;
  const int lane = tid & 31, wave = tid >> 5;
  const int half = lane >> 4, l16 = lane & 15;

  for (int i = tid; i < 784; i += 256) xs[i] = x[(size_t)b * 784 + i];
  for (int i = tid; i < 32 * 64; i += 256) {
    int n = i >> 6, k = i & 63;
    float w = enc1_w[min(n, K1C - 1) * 64 + k];
    As[0] = As[0];  // no-op
    Bs[i] = (_Float16)((n < K1C) ? w : 0.f);
  }
  for (int i = tid; i < 162; i += 256) pw[i] = (i < 81) ? pw1[i] : pw2[i - 81];
  __syncthreads();

  // im2col fill (rows 441..447 duplicate row 440; discarded later)
  for (int idx = tid; idx < 448 * 64; idx += 256) {
    int m = idx >> 6, k = idx & 63;
    int p = min(m, 440);
    int oy = p / 21, ox = p - oy * 21;
    As[idx] = (_Float16)xs[(oy + (k >> 3)) * 28 + ox + (k & 7)];
  }
  __syncthreads();

  // GEMM: 28 M-tiles x 2 N-tiles, K=64 (2 wmma), pure b128 DS + wmma
  for (int mt = wave; mt < 28; mt += 8) {
    for (int nt = 0; nt < 2; ++nt) {
      v8f acc = {};
#pragma unroll
      for (int ks = 0; ks < 2; ++ks) {
        v16h Af = ldA(As, 64, mt * 16 + l16, ks * 32, half);
        v16h Bf = ldB(Bs, 64, nt * 16 + l16, ks * 32, half);
        acc = WMMA_F32_F16(Af, Bf, acc);
      }
#pragma unroll
      for (int r = 0; r < 8; ++r)
        c1[(mt * 16 + r + 8 * half) * 32 + nt * 16 + l16] = acc[r];
    }
  }
  __syncthreads();

  // stochastic pooling over 3x3 tiles
  for (int idx = tid; idx < K1C * 49; idx += 256) {
    int c = idx / 49, nm = idx % 49, n = nm / 7, m = nm % 7;
    float tile[9];
#pragma unroll
    for (int t = 0; t < 9; ++t)
      tile[t] = c1[((n * 3 + t / 3) * 21 + (m * 3 + t % 3)) * 32 + c];
    float h1[9];
#pragma unroll
    for (int o = 0; o < 9; ++o) {
      float s = 0.f;
#pragma unroll
      for (int t = 0; t < 9; ++t) s += pw[o * 9 + t] * tile[t];
      h1[o] = tanhf(s);
    }
    float e[9], mx = -1e30f;
#pragma unroll
    for (int o = 0; o < 9; ++o) {
      float s = 0.f;
#pragma unroll
      for (int t = 0; t < 9; ++t) s += pw[81 + o * 9 + t] * h1[t];
      e[o] = s; mx = fmaxf(mx, s);
    }
    float den = 0.f;
#pragma unroll
    for (int o = 0; o < 9; ++o) { e[o] = __expf(e[o] - mx); den += e[o]; }
    float inv = 1.f / den;
    float* ep = eta_out + ((size_t)(b * K1C + c) * 49 + nm) * 9;
#pragma unroll
    for (int o = 0; o < 9; ++o) { e[o] *= inv; ep[o] = e[o]; }
    float u = u01(hash3((uint32_t)b, (uint32_t)idx, 0x706F6F6Cu));
    int z = 8; float cs = 0.f;
    for (int o = 0; o < 9; ++o) { cs += e[o]; if (u < cs) { z = o; break; } }
    pooled[(size_t)(b * K1C + c) * 49 + nm] = tile[z];
  }
}

// =====================================================================
// Kernel 2: enc2 conv4x4, 8 images/block (M=128), N=80, K=480.
// f16 weight panel staged once in LDS, reused by all 8 waves (wave=image).
// Dynamic LDS: As 122880 | Bs 76800 = 199680 B (320KB WGP budget).
// =====================================================================
__global__ __launch_bounds__(256) void enc2_kernel(
    const float* __restrict__ pooled, const float* __restrict__ enc2_w,
    float* __restrict__ flat) {
  _Float16* As = (_Float16*)smem_raw;              // [128][480]
  _Float16* Bs = (_Float16*)(smem_raw + 122880);   // [80][480] (row-major == enc2_w)

  const int b0 = blockIdx.x * 8, tid = threadIdx.x;
  for (int i = tid; i < K2C * 480; i += 256) Bs[i] = (_Float16)enc2_w[i];
  for (int idx = tid; idx < 128 * 480; idx += 256) {
    int m = idx / 480, k = idx - m * 480;
    int img = m >> 4, pos = m & 15;
    int c = k >> 4, r = k & 15;
    As[idx] = (_Float16)pooled[((size_t)(b0 + img) * K1C + c) * 49 +
                               ((pos >> 2) + (r >> 2)) * 7 + (pos & 3) + (r & 3)];
  }
  __syncthreads();

  const int lane = tid & 31, wave = tid >> 5;   // wave == image
  const int half = lane >> 4, l16 = lane & 15;
  const int m = wave * 16 + l16;
  for (int nt = 0; nt < 5; ++nt) {
    v8f acc = {};
#pragma unroll 5
    for (int ks = 0; ks < 15; ++ks) {
      v16h Af = ldA(As, 480, m, ks * 32, half);
      v16h Bf = ldB(Bs, 480, nt * 16 + l16, ks * 32, half);
      acc = WMMA_F32_F16(Af, Bf, acc);
    }
    int o = nt * 16 + l16;
#pragma unroll
    for (int r = 0; r < 8; ++r)
      flat[((size_t)(b0 + wave) * K2C + o) * 16 + r + 8 * half] = acc[r];
  }
}

// =====================================================================
// Kernel 3: per-channel code MLPs as batch-tiled 16x16 GEMMs (1 wave).
// Zero-padded K handled at fill time -> hot path is pure b128 DS + wmma.
// =====================================================================
__global__ __launch_bounds__(32) void code_mlp_kernel(
    const float* __restrict__ flat,
    const float* __restrict__ h_w, const float* __restrict__ h_b,
    const float* __restrict__ mean_w, const float* __restrict__ mean_b,
    const float* __restrict__ logvar_w, const float* __restrict__ logvar_b,
    float* __restrict__ mean_out, float* __restrict__ logvar_out,
    float* __restrict__ s_out) {
  __shared__ _Float16 Ap[16 * 32], Hp[16 * 32];
  __shared__ _Float16 Bh[16 * 32], Bm[16 * 32], Bl[16 * 32];
  const int kch = blockIdx.x, b0 = blockIdx.y * 16;
  const int lane = threadIdx.x, half = lane >> 4, l16 = lane & 15;

  for (int idx = lane; idx < 512; idx += 32) {
    int mn = idx >> 5, k = idx & 31;
    int kc = min(k, 15);
    float a  = flat[((size_t)(b0 + mn) * K2C + kch) * 16 + kc];
    float bh = h_w[((size_t)kch * 16 + mn) * 16 + kc];
    float bm = mean_w[((size_t)kch * 16 + mn) * 16 + kc];
    float bl = logvar_w[((size_t)kch * 16 + mn) * 16 + kc];
    bool z = (k < 16);
    Ap[idx] = (_Float16)(z ? a : 0.f);
    Bh[idx] = (_Float16)(z ? bh : 0.f);
    Bm[idx] = (_Float16)(z ? bm : 0.f);
    Bl[idx] = (_Float16)(z ? bl : 0.f);
  }
  __syncthreads();

  v8f ah = {};
  ah = WMMA_F32_F16(ldA(Ap, 32, l16, 0, half), ldB(Bh, 32, l16, 0, half), ah);
#pragma unroll
  for (int r = 0; r < 8; ++r) {
    int mrow = r + 8 * half;
    Hp[mrow * 32 + l16]      = (_Float16)tanhf(ah[r] + h_b[kch * 16 + l16]);
    Hp[mrow * 32 + 16 + l16] = (_Float16)0.f;
  }
  __syncthreads();

  v16h Ah = ldA(Hp, 32, l16, 0, half);
  v8f am = {}, al = {};
  am = WMMA_F32_F16(Ah, ldB(Bm, 32, l16, 0, half), am);
  al = WMMA_F32_F16(Ah, ldB(Bl, 32, l16, 0, half), al);
#pragma unroll
  for (int r = 0; r < 8; ++r) {
    int bg = b0 + r + 8 * half, o = l16;
    float mv = am[r] + mean_b[kch * 16 + o];
    float lv = al[r] + logvar_b[kch * 16 + o];
    size_t off = ((size_t)bg * K2C + kch) * 16 + o;
    mean_out[off]   = mv;
    logvar_out[off] = lv;
    float g = gaussf((uint32_t)bg, (uint32_t)(kch * 16 + o), 0x636F6465u);
    s_out[off] = mv + __expf(0.5f * lv) * g;
  }
}

// =====================================================================
// Kernel 4: dec2 convT (M=49->64, N=30->32, K=1280, A chunked 5x256) +
// unpool + dec1 convT (VALU matvec) + diagonal noise. Weight panel
// pre-flipped to f16 in LDS ((3-kh)*4+(3-kw) == 15-r).
// Dynamic LDS: ss 5120 | s2 8192 | Ac 32768 | Bs 81920 | s1 52928 | w1 7680
//            = 188608 B
// =====================================================================
__global__ __launch_bounds__(256) void dec_kernel(
    const float* __restrict__ s_in, const float* __restrict__ dec2_w,
    const float* __restrict__ dec1_w, const float* __restrict__ eta,
    const float* __restrict__ alpha_p, float* __restrict__ xrec) {
  float*    ss = (float*)smem_raw;                       // [80][16]
  float*    s2 = (float*)(smem_raw + 5120);              // [64][32]
  _Float16* Ac = (_Float16*)(smem_raw + 13312);          // [64][256]
  _Float16* Bs = (_Float16*)(smem_raw + 46080);          // [32][1280] pre-flipped
  float*    s1 = (float*)(smem_raw + 128000);            // [30][441]
  float*    w1 = (float*)(smem_raw + 180928);            // [1920]

  const int b = blockIdx.x, tid = threadIdx.x;
  for (int i = tid; i < K2C * 16; i += 256) ss[i] = s_in[(size_t)b * K2C * 16 + i];
  for (int i = tid; i < 1920; i += 256) w1[i] = dec1_w[i];
  for (int idx = tid; idx < 32 * 1280; idx += 256) {
    int n = idx / 1280, K = idx - n * 1280;
    int i = K >> 4, r = K & 15;
    float w = dec2_w[((size_t)i * K1C + min(n, K1C - 1)) * 16 + 15 - r];
    Bs[idx] = (_Float16)((n < K1C) ? w : 0.f);
  }
  __syncthreads();

  const int lane = tid & 31, wave = tid >> 5;
  const int half = lane >> 4, l16 = lane & 15;
  const int mtile = wave >> 1, nt = wave & 1;
  const int m = mtile * 16 + l16, n = nt * 16 + l16;

  v8f acc = {};
  for (int chnk = 0; chnk < 5; ++chnk) {
    __syncthreads();
    for (int idx = tid; idx < 64 * 256; idx += 256) {
      int mm = idx >> 8, kk = idx & 255;
      int i = chnk * 16 + (kk >> 4), r = kk & 15;
      int pos = min(mm, 48);
      int y = pos / 7, xx = pos - y * 7;
      int u = y + (r >> 2) - 3, v = xx + (r & 3) - 3;
      float val = ss[i * 16 + (u & 3) * 4 + (v & 3)];          // unconditional load
      bool inb = (u >= 0) & (u < 4) & (v >= 0) & (v < 4) & (mm < 49);
      Ac[idx] = (_Float16)(inb ? val : 0.f);                   // cndmask, no branch
    }
    __syncthreads();
#pragma unroll 4
    for (int ks = 0; ks < 8; ++ks) {
      v16h Af = ldA(Ac, 256, m, ks * 32, half);
      v16h Bf = ldB(Bs, 1280, n, chnk * 256 + ks * 32, half);
      acc = WMMA_F32_F16(Af, Bf, acc);
    }
  }
#pragma unroll
  for (int r = 0; r < 8; ++r)
    s2[(mtile * 16 + r + 8 * half) * 32 + n] = acc[r];
  __syncthreads();

  for (int i = tid; i < K1C * 441; i += 256) s1[i] = 0.f;
  __syncthreads();
  for (int idx = tid; idx < K1C * 49; idx += 256) {
    int c = idx / 49, nm = idx % 49, nn = nm / 7, mm = nm % 7;
    const float* ep = eta + ((size_t)(b * K1C + c) * 49 + nm) * 9;
    float u = u01(hash3((uint32_t)b, (uint32_t)idx, 0x756E706Cu));
    int z = 8; float cs = 0.f;
    for (int t = 0; t < 9; ++t) { cs += ep[t]; if (u < cs) { z = t; break; } }
    s1[c * 441 + (nn * 3 + z / 3) * 21 + (mm * 3 + z % 3)] = s2[nm * 32 + c];
  }
  __syncthreads();

  const float inva = 1.0f / alpha_p[0];
  for (int p = tid; p < 784; p += 256) {
    int i = p / 28, j = p - 28 * (p / 28);
    int khlo = max(0, 7 - i), khhi = min(8, 28 - i);
    int kwlo = max(0, 7 - j), kwhi = min(8, 28 - j);
    float a1 = 0.f;
    for (int c = 0; c < K1C; ++c) {
      const float* s1c = s1 + c * 441;
      const float* wc  = w1 + c * 64;
      for (int kh = khlo; kh < khhi; ++kh) {
        int u = i + kh - 7;
        for (int kw = kwlo; kw < kwhi; ++kw)
          a1 += s1c[u * 21 + j + kw - 7] * wc[(7 - kh) * 8 + 7 - kw];
      }
    }
    if (i == j) a1 += inva * gaussf((uint32_t)b, (uint32_t)p, 0x6E6F6973u);
    xrec[(size_t)b * 784 + p] = a1;
  }
}

// =====================================================================
extern "C" void kernel_launch(void* const* d_in, const int* in_sizes, int n_in,
                              void* d_out, int out_size, void* d_ws, size_t ws_size,
                              hipStream_t stream) {
  (void)in_sizes; (void)n_in; (void)out_size; (void)ws_size;
  const float* x        = (const float*)d_in[0];
  const float* enc1_w   = (const float*)d_in[1];
  const float* pool_w1  = (const float*)d_in[2];
  const float* pool_w2  = (const float*)d_in[3];
  const float* enc2_w   = (const float*)d_in[4];
  const float* h_w      = (const float*)d_in[5];
  const float* h_b      = (const float*)d_in[6];
  const float* mean_w   = (const float*)d_in[7];
  const float* mean_b   = (const float*)d_in[8];
  const float* logvar_w = (const float*)d_in[9];
  const float* logvar_b = (const float*)d_in[10];
  const float* dec2_w   = (const float*)d_in[11];
  const float* dec1_w   = (const float*)d_in[12];
  const float* alpha    = (const float*)d_in[13];

  // d_out = [x_rec | mean | log_var | eta]
  float* xrec       = (float*)d_out;
  float* mean_out   = xrec + (size_t)BATCH * 784;
  float* logvar_out = mean_out + (size_t)BATCH * K2C * 16;
  float* eta_out    = logvar_out + (size_t)BATCH * K2C * 16;

  float* ws     = (float*)d_ws;
  float* pooled = ws;                                   // [B,30,49]
  float* flat   = pooled + (size_t)BATCH * K1C * 49;    // [B,80,16]
  float* s_buf  = flat + (size_t)BATCH * K2C * 16;      // [B,80,16]

  enc1_pool_kernel<<<BATCH, 256, 122568, stream>>>(x, enc1_w, pool_w1, pool_w2,
                                                   pooled, eta_out);
  enc2_kernel<<<BATCH / 8, 256, 199680, stream>>>(pooled, enc2_w, flat);
  dim3 g3(K2C, BATCH / 16);
  code_mlp_kernel<<<g3, 32, 0, stream>>>(flat, h_w, h_b, mean_w, mean_b,
                                         logvar_w, logvar_b,
                                         mean_out, logvar_out, s_buf);
  dec_kernel<<<BATCH, 256, 188608, stream>>>(s_buf, dec2_w, dec1_w, eta_out,
                                             alpha, xrec);
}
